// SpatialCrossAttention_22196390985993
// MI455X (gfx1250) — matrix operand order repaired and checked by reading, not compiled, not run on hardware
//
#include <hip/hip_runtime.h>
#include <hip/hip_fp16.h>
#include <stdint.h>

typedef __attribute__((ext_vector_type(16))) _Float16 v16h;
typedef __attribute__((ext_vector_type(8)))  _Float16 v8h;
typedef __attribute__((ext_vector_type(8)))  float    v8f;

#define NQ     4096
#define NCAM   6
#define NV     22050
#define EMBED  256
#define KDIM   256
#define NHEAD  8
#define NLVL   3
#define NPTS   8
#define NPAR   576   // 384 offsets + 192 attention logits

// ---------------------------------------------------------------------------
// CDNA5 async global->LDS copy (ASYNCcnt-tracked) helpers
// ---------------------------------------------------------------------------
__device__ __forceinline__ void async_copy_b128(_Float16* lds_dst, const _Float16* gsrc) {
    uint32_t l = (uint32_t)(uintptr_t)lds_dst;           // LDS addr = generic addr[31:0]
    uint64_t g = (uint64_t)(uintptr_t)gsrc;
    asm volatile("global_load_async_to_lds_b128 %0, %1, off"
                 :: "v"(l), "v"(g) : "memory");
}
__device__ __forceinline__ void wait_async0() {
    asm volatile("s_wait_asynccnt 0" ::: "memory");
}

__device__ __forceinline__ v16h pack_a_frag(const _Float16* p) {
    v8h lo = *reinterpret_cast<const v8h*>(p);
    v8h hi = *reinterpret_cast<const v8h*>(p + 16);
    return __builtin_shufflevector(lo, hi, 0,1,2,3,4,5,6,7,8,9,10,11,12,13,14,15);
}
__device__ __forceinline__ v8f wmma16(v16h a, v16h b, v8f c) {
    return __builtin_amdgcn_wmma_f32_16x16x32_f16(false, a, false, b,
                                                  (short)0, c, false, false);
}

// ---------------------------------------------------------------------------
// Pack: qsum = query + query_pos (f32); weights -> f16; fused bias vector
// ---------------------------------------------------------------------------
__global__ void pack_kernel(const float* __restrict__ query,
                            const float* __restrict__ qpos,
                            const float* __restrict__ Wv,
                            const float* __restrict__ Ws,
                            const float* __restrict__ Wa,
                            const float* __restrict__ Wo,
                            const float* __restrict__ b_off,
                            const float* __restrict__ ba,
                            float* __restrict__ qsum,
                            _Float16* __restrict__ Wv16,
                            _Float16* __restrict__ Wsa16,
                            _Float16* __restrict__ Wo16,
                            float* __restrict__ bias_sa)
{
    int idx = blockIdx.x * blockDim.x + threadIdx.x;
    if (idx < NQ * EMBED)       qsum[idx] = query[idx] + qpos[idx];
    if (idx < EMBED * EMBED) {  Wv16[idx] = (_Float16)Wv[idx];
                                Wo16[idx] = (_Float16)Wo[idx]; }
    if (idx < EMBED * NPAR) {
        int k = idx / NPAR, n = idx % NPAR;
        Wsa16[idx] = (_Float16)((n < 384) ? Ws[k * 384 + n] : Wa[k * 192 + (n - 384)]);
    }
    if (idx < NPAR) bias_sa[idx] = (idx < 384) ? b_off[idx] : ba[idx - 384];
}

// ---------------------------------------------------------------------------
// WMMA GEMM: C[M,N] = A(f32)[M,K=256] * B(f16)[K,N] + bias
//   MODE 0: store f16 (value projection)
//   MODE 1: store f32 (sampling params, raw)
//   MODE 2: store f32 + residual (output projection)
// Block: 256 threads = 8 waves (4x2). Tile 128(M) x 64(N), K-step 32,
// double-buffered LDS. Each wave: 32x32 register tile = 4 v_wmma per step.
// A staged via f32 loads + cvt + ds_store; B staged async-to-LDS.
// ---------------------------------------------------------------------------
template<int MODE>
__global__ __launch_bounds__(256)
void wmma_gemm(const float* __restrict__ A, const _Float16* __restrict__ B,
               const float* __restrict__ bias, const float* __restrict__ resid,
               void* __restrict__ outp, int M, int N)
{
    __shared__ __align__(32) _Float16 sA[2][128 * 32];
    __shared__ __align__(32) _Float16 sB[2][32 * 64];

    const int tid  = threadIdx.x;
    const int lane = tid & 31;
    const int wave = tid >> 5;
    const int wm   = wave >> 1;         // 0..3 -> 32-row sub-tile
    const int wn   = wave & 1;          // 0..1 -> 32-col sub-tile
    const int m0   = blockIdx.y * 128;
    const int n0   = blockIdx.x * 64;

    // A staging: 128 rows x 32 halves; each thread covers 16 halves of one row
    const int arw = tid >> 1;
    const int acl = (tid & 1) * 16;
    int gar = m0 + arw; if (gar >= M) gar = M - 1;          // clamp tail rows
    const float* aptr = A + (size_t)gar * KDIM + acl;

    // B staging: 32 rows x 64 halves; each thread one async b128 (8 halves)
    const int brw = tid >> 3;
    const int bcl = (tid & 7) * 8;

    v8f c00 = {}, c01 = {}, c10 = {}, c11 = {};

    // fragment addressing (per documented 16-bit WMMA VGPR layouts)
    const int ar  = wm * 32 + (lane & 15);
    const int akb = (lane < 16) ? 0 : 8;
    const int bkr = (lane & 15) + ((lane < 16) ? 0 : 16);
    const int bnc = wn * 32;

    auto stage = [&](int b, int k0) {
        async_copy_b128(&sB[b][brw * 64 + bcl],
                        B + (size_t)(k0 + brw) * N + n0 + bcl);
        const float* ap = aptr + k0;
        float4 f0 = *reinterpret_cast<const float4*>(ap);
        float4 f1 = *reinterpret_cast<const float4*>(ap + 4);
        float4 f2 = *reinterpret_cast<const float4*>(ap + 8);
        float4 f3 = *reinterpret_cast<const float4*>(ap + 12);
        v8h h0 = { (_Float16)f0.x, (_Float16)f0.y, (_Float16)f0.z, (_Float16)f0.w,
                   (_Float16)f1.x, (_Float16)f1.y, (_Float16)f1.z, (_Float16)f1.w };
        v8h h1 = { (_Float16)f2.x, (_Float16)f2.y, (_Float16)f2.z, (_Float16)f2.w,
                   (_Float16)f3.x, (_Float16)f3.y, (_Float16)f3.z, (_Float16)f3.w };
        *reinterpret_cast<v8h*>(&sA[b][arw * 32 + acl])     = h0;
        *reinterpret_cast<v8h*>(&sA[b][arw * 32 + acl + 8]) = h1;
        if (k0 + 32 < KDIM) __builtin_prefetch(aptr + k0 + 32);  // global_prefetch_b8
    };
    auto compute = [&](int b) {
        v16h a0 = pack_a_frag(&sA[b][ ar       * 32 + akb]);
        v16h a1 = pack_a_frag(&sA[b][(ar + 16) * 32 + akb]);
        v16h b0, b1;
        #pragma unroll
        for (int s = 0; s < 16; ++s) {
            const int kk = akb + (s & 7) + ((s >> 3) << 4);
            b0[s] = sB[b][kk * 64 + bnc + (lane & 15)];
            b1[s] = sB[b][kk * 64 + bnc + 16 + (lane & 15)];
        }
        c00 = wmma16(a0, b0, c00);
        c01 = wmma16(a0, b1, c01);
        c10 = wmma16(a1, b0, c10);
        c11 = wmma16(a1, b1, c11);
    };

    // software pipeline: stage k+1 while computing k (double-buffered LDS)
    int buf = 0;
    stage(0, 0);
    wait_async0();
    __syncthreads();
    for (int k0 = 32; k0 < KDIM; k0 += 32) {
        stage(buf ^ 1, k0);
        compute(buf);
        wait_async0();
        __syncthreads();
        buf ^= 1;
    }
    compute(buf);

    // C layout: VGPR i -> M = i + (lane<16 ? 0 : 8), N = lane&15
    const int lrow = (lane < 16) ? 0 : 8;
    const int lcol = lane & 15;
    #pragma unroll
    for (int tm = 0; tm < 2; ++tm) {
        #pragma unroll
        for (int tn = 0; tn < 2; ++tn) {
            const v8f& c = tm ? (tn ? c11 : c10) : (tn ? c01 : c00);
            const int rbase = m0 + wm * 32 + tm * 16 + lrow;
            const int gcol  = n0 + wn * 32 + tn * 16 + lcol;
            #pragma unroll
            for (int i = 0; i < 8; ++i) {
                int gr = rbase + i;
                if (gr < M) {
                    float v = c[i] + bias[gcol];
                    if constexpr (MODE == 2) v += resid[(size_t)gr * N + gcol];
                    if constexpr (MODE == 0)
                        ((_Float16*)outp)[(size_t)gr * N + gcol] = (_Float16)v;
                    else
                        ((float*)outp)[(size_t)gr * N + gcol] = v;
                }
            }
        }
    }
}

// ---------------------------------------------------------------------------
// Per-(q,head): softmax over 24 logits, scale offsets by 1/[w,h] per level.
// ---------------------------------------------------------------------------
__global__ void finalize_params(float* __restrict__ praw, float* __restrict__ aw)
{
    int idx = blockIdx.x * blockDim.x + threadIdx.x;
    if (idx >= NQ * NHEAD) return;
    int q = idx >> 3, h = idx & 7;
    float* row = praw + (size_t)q * NPAR;

    float lg[24], m = -1e30f;
    #pragma unroll
    for (int j = 0; j < 24; ++j) { lg[j] = row[384 + h * 24 + j]; m = fmaxf(m, lg[j]); }
    float s = 0.f;
    #pragma unroll
    for (int j = 0; j < 24; ++j) { lg[j] = __expf(lg[j] - m); s += lg[j]; }
    float inv = 1.f / s;
    #pragma unroll
    for (int j = 0; j < 24; ++j) aw[((size_t)q * NHEAD + h) * 24 + j] = lg[j] * inv;

    const float iw[3] = { 1.f/168.f, 1.f/84.f, 1.f/42.f };
    const float ih[3] = { 1.f/100.f, 1.f/50.f, 1.f/25.f };
    #pragma unroll
    for (int l = 0; l < NLVL; ++l)
        for (int p = 0; p < NPTS; ++p) {
            int o = ((h * NLVL + l) * NPTS + p) * 2;
            row[o]     *= iw[l];
            row[o + 1] *= ih[l];
        }
}

// ---------------------------------------------------------------------------
// Deformable bilinear sampling + cross-camera hit-masked reduction.
// One wave32 per (query, head); lane = channel. Coalesced 64B corner reads
// out of L2-resident f16 projected values.
// ---------------------------------------------------------------------------
__global__ __launch_bounds__(256)
void sample_kernel(const _Float16* __restrict__ val16,
                   const float* __restrict__ off,   // praw cols 0..383 (scaled)
                   const float* __restrict__ aw,
                   const float* __restrict__ ref,   // (6,1,4096,4,2)
                   const unsigned char* __restrict__ mask, // (6,1,4096,4) bool
                   float* __restrict__ slots)
{
    const int q    = blockIdx.x;
    const int h    = threadIdx.x >> 5;
    const int lane = threadIdx.x & 31;
    const int LST[3] = { 0, 16800, 21000 };
    const int LW [3] = { 168, 84, 42 };
    const int LH [3] = { 100, 50, 25 };

    float num = 0.f, cnt = 0.f;
    const float* orow = off + (size_t)q * NPAR;
    const float* arow = aw  + ((size_t)q * NHEAD + h) * 24;

    for (int cam = 0; cam < NCAM; ++cam) {
        const unsigned char* mk = mask + ((size_t)cam * NQ + q) * 4;
        bool hit = (mk[0] | mk[1] | mk[2] | mk[3]) != 0;
        const float* rf = ref + ((size_t)cam * NQ + q) * 8;
        float o = 0.f;
        for (int l = 0; l < NLVL; ++l) {
            const int W = LW[l], H = LH[l];
            const _Float16* vb = val16
                + ((size_t)(cam * NV + LST[l])) * EMBED + h * 32 + lane;
            #pragma unroll
            for (int p = 0; p < NPTS; ++p) {
                int an = p & 3;
                int oi = ((h * NLVL + l) * NPTS + p) * 2;
                float lx = rf[an * 2]     + orow[oi];
                float ly = rf[an * 2 + 1] + orow[oi + 1];
                float x = lx * (float)W - 0.5f;
                float y = ly * (float)H - 0.5f;
                float xf = floorf(x), yf = floorf(y);
                int   x0 = (int)xf,   y0 = (int)yf;
                float fx = x - xf,    fy = y - yf;
                float s = 0.f;
                #pragma unroll
                for (int dy = 0; dy < 2; ++dy)
                    #pragma unroll
                    for (int dx = 0; dx < 2; ++dx) {
                        int xi = x0 + dx, yi = y0 + dy;
                        if (xi >= 0 && xi < W && yi >= 0 && yi < H) {
                            float wgt = (dx ? fx : 1.f - fx) * (dy ? fy : 1.f - fy);
                            s += wgt * (float)vb[(size_t)(yi * W + xi) * EMBED];
                        }
                    }
                o += s * arow[l * NPTS + p];
            }
        }
        if (hit) { num += o; cnt += 1.f; }
    }
    slots[(size_t)q * EMBED + h * 32 + lane] = num / fmaxf(cnt, 1.f);
}

// ---------------------------------------------------------------------------
extern "C" void kernel_launch(void* const* d_in, const int* in_sizes, int n_in,
                              void* d_out, int out_size, void* d_ws, size_t ws_size,
                              hipStream_t stream) {
    const float* query = (const float*)d_in[0];
    const float* value = (const float*)d_in[2];
    const float* qpos  = (const float*)d_in[3];
    const float* ref   = (const float*)d_in[4];
    const unsigned char* bmask = (const unsigned char*)d_in[5];
    const float* Wv    = (const float*)d_in[8];
    const float* bv    = (const float*)d_in[9];
    const float* Ws    = (const float*)d_in[10];
    const float* b_off = (const float*)d_in[11];
    const float* Wa    = (const float*)d_in[12];
    const float* ba    = (const float*)d_in[13];
    const float* Wo    = (const float*)d_in[14];
    const float* bo    = (const float*)d_in[15];

    // workspace carve-out
    char*  base = (char*)d_ws;
    size_t off  = 0;
    auto carve = [&](size_t bytes) { char* p = base + off;
                                     off = (off + bytes + 255) & ~(size_t)255; return p; };
    float*    qsum   = (float*)   carve((size_t)NQ * EMBED * 4);
    _Float16* Wv16   = (_Float16*)carve((size_t)EMBED * EMBED * 2);
    _Float16* Wsa16  = (_Float16*)carve((size_t)EMBED * NPAR * 2);
    _Float16* Wo16   = (_Float16*)carve((size_t)EMBED * EMBED * 2);
    float*    biassa = (float*)   carve((size_t)NPAR * 4);
    float*    praw   = (float*)   carve((size_t)NQ * NPAR * 4);
    float*    aw     = (float*)   carve((size_t)NQ * NHEAD * 24 * 4);
    _Float16* val16  = (_Float16*)carve((size_t)NCAM * NV * EMBED * 2);
    float*    slots  = (float*)   carve((size_t)NQ * EMBED * 4);

    // 1) pack
    pack_kernel<<<(NQ * EMBED + 255) / 256, 256, 0, stream>>>(
        query, qpos, Wv, Ws, Wa, Wo, b_off, ba, qsum, Wv16, Wsa16, Wo16, biassa);

    // 2) sampling-parameter GEMM (q broadcast across cameras -> compute once)
    wmma_gemm<1><<<dim3(NPAR / 64, NQ / 128), 256, 0, stream>>>(
        qsum, Wsa16, biassa, nullptr, (void*)praw, NQ, NPAR);

    // 3) softmax + offset normalization
    finalize_params<<<(NQ * NHEAD + 255) / 256, 256, 0, stream>>>(praw, aw);

    // 4) value projection GEMM (dominant): M = 6*22050, f16 output
    const int Mval = NCAM * NV;
    wmma_gemm<0><<<dim3(EMBED / 64, (Mval + 127) / 128), 256, 0, stream>>>(
        value, Wv16, bv, nullptr, (void*)val16, Mval, EMBED);

    // 5) deformable sampling + cross-camera reduction
    sample_kernel<<<NQ, 256, 0, stream>>>(val16, praw, aw, ref, bmask, slots);

    // 6) output projection + bias + residual
    wmma_gemm<2><<<dim3(EMBED / 64, NQ / 128), 256, 0, stream>>>(
        slots, Wo16, bo, query, d_out, NQ, EMBED);
}
